// HM_LSTM_Actor_26010321945098
// MI455X (gfx1250) — matrix operand order repaired and checked by reading, not compiled, not run on hardware
//
#include <hip/hip_runtime.h>
#include <hip/hip_bf16.h>
#include <stdint.h>

typedef __attribute__((ext_vector_type(16))) __bf16 v16bf;
typedef __attribute__((ext_vector_type(8)))  float  v8f;
typedef __attribute__((ext_vector_type(4)))  int    v4i;

#define Tt   256
#define Bsz  1024
#define HH   256
#define ACTN 64
#define TB   (Tt * Bsz)

__device__ __forceinline__ unsigned short f2bf(float f) {
  union { float f; unsigned int u; } c; c.f = f;
  unsigned int u = c.u;
  u += 0x7FFFu + ((u >> 16) & 1u);
  return (unsigned short)(u >> 16);
}
__device__ __forceinline__ float bf2f(unsigned short b) {
  union { unsigned int u; float f; } c; c.u = ((unsigned int)b) << 16;
  return c.f;
}
__device__ __forceinline__ float sigmoidf_(float x) {
  return 1.0f / (1.0f + __expf(-x));
}
__device__ __forceinline__ unsigned int lds_off(const void* p) {
  return (unsigned int)(size_t)p;   // low 32 bits of LDS flat address = LDS offset
}

struct FuseArgs {
  float* c;                  // cell state (RMW, element-owned)
  unsigned short* hbf_out;   // new h (bf16), double-buffered across t
  unsigned short* zh_out;    // z * h_new (bf16) -> next GEMM's A chunk
  float* z1;                 // per-row z (cell1 writes, cell2 reads)
  const float* noise;        // noise1[t] (cell1)
  const float* zw;           // stacked z-column weights [K] (cell1)
  const float* bz;           // z-column bias (cell1)
  unsigned short* emb;       // emb[t] output (cell2)
};

// ---------------------------------------------------------------------------
// WMMA GEMM over a chunked A operand: A = [A0 | A1c | A2c], each [M,256] bf16
// (A0 may be f32 when AF32). W: bf16 [N][K] pre-transposed (K contiguous),
// rows gate-interleaved for EPI>=3.
//   EPI 1: bf16 relu out      EPI 2: f32 tanh out
//   EPI 3: fused HM-LSTM cell-1 update epilogue (no C written)
//   EPI 4: fused HM-LSTM cell-2 update epilogue (no C written)
// Tiling: WG 64(M) x 128(N), 8 waves (4Mx2N), wave tile 16x64, K step 32.
// Staging: GLOBAL_LOAD_ASYNC_TO_LDS_B128, double-buffered, ASYNCcnt pipelined.
// ---------------------------------------------------------------------------
template<int EPI, bool AF32>
__global__ __launch_bounds__(256) void gemm_wmma(
    const void* __restrict__ A0, const unsigned short* __restrict__ A1c,
    const unsigned short* __restrict__ A2c, const unsigned short* __restrict__ W,
    const float* __restrict__ bias, void* __restrict__ Cptr,
    int M, int N, int K, FuseArgs fa)
{
  __shared__ __align__(16) unsigned short As[2][64 * 40];
  __shared__ __align__(16) unsigned short Wst[2][128 * 40];
  const int tid  = threadIdx.x;
  const int lane = tid & 31;
  const int wave = tid >> 5;
  const int wm   = wave & 3;
  const int wn   = wave >> 2;
  const int mBlock = blockIdx.x * 64;
  const int nBlock = blockIdx.y * 128;

  v8f acc[4];
  #pragma unroll
  for (int t = 0; t < 4; ++t)
    #pragma unroll
    for (int i = 0; i < 8; ++i) acc[t][i] = 0.0f;

  const int laneRow = lane & 15;
  const int hiHalf  = lane >> 4;

  const int rA  = tid >> 2;          // 0..63
  const int kcA = (tid & 3) * 8;     // 0,8,16,24
  const int rW  = tid >> 1;          // 0..127
  const int kcW = (tid & 1) * 16;    // 0 or 16
  int nrow = nBlock + rW;
  if (nrow > N - 1) nrow = N - 1;    // clamp; garbage cols dropped in epilogue

  auto issue = [&](int buf, int k0) {
    if (!AF32) {
      int chunk = k0 >> 8;
      const unsigned short* ab = (chunk == 0) ? (const unsigned short*)A0
                               : (chunk == 1) ? A1c : A2c;
      unsigned int la = lds_off(&As[buf][rA * 40 + kcA]);
      const unsigned short* ga = ab + (long)(mBlock + rA) * 256 + (k0 & 255) + kcA;
      asm volatile("global_load_async_to_lds_b128 %0, %1, off"
                   :: "v"(la), "v"(ga) : "memory");
    }
    unsigned int lw0 = lds_off(&Wst[buf][rW * 40 + kcW]);
    const unsigned short* gw = W + (long)nrow * K + k0 + kcW;
    asm volatile("global_load_async_to_lds_b128 %0, %1, off"
                 :: "v"(lw0), "v"(gw) : "memory");
    asm volatile("global_load_async_to_lds_b128 %0, %1, off"
                 :: "v"(lw0 + 16u), "v"(gw + 8) : "memory");
  };
  auto stageA_f32 = [&](int buf, int k0) {
    const float* src = (const float*)A0 + (long)(mBlock + rA) * 256 + k0 + kcA;
    __align__(16) unsigned short t8[8];
    #pragma unroll
    for (int j = 0; j < 8; ++j) t8[j] = f2bf(src[j]);
    *(v4i*)&As[buf][rA * 40 + kcA] = *(const v4i*)t8;
  };

  issue(0, 0);
  if (AF32) stageA_f32(0, 0);

  float zsum = 0.0f;                 // cell1: z-column partial (4 lanes per row)
  const int nsteps = K >> 5;
  for (int it = 0; it < nsteps; ++it) {
    const int buf = it & 1;
    const bool more = (it + 1 < nsteps);
    if (more) {
      issue(buf ^ 1, (it + 1) << 5);
      if (AF32) stageA_f32(buf ^ 1, (it + 1) << 5);
      if (AF32) asm volatile("s_wait_asynccnt 2" ::: "memory");
      else      asm volatile("s_wait_asynccnt 3" ::: "memory");
    } else {
      asm volatile("s_wait_asynccnt 0" ::: "memory");
    }
    __syncthreads();

    if constexpr (EPI == 3) {        // fused z-column dot from staged A tile
      const float* zwp = fa.zw + (it << 5) + kcA;
      const unsigned short* asrc = &As[buf][rA * 40 + kcA];
      #pragma unroll
      for (int j = 0; j < 8; ++j) zsum += bf2f(asrc[j]) * zwp[j];
    }

    // load A fragment and ALL four B fragments first (distinct regs), then
    // issue the 4 WMMAs back-to-back (no D->A/B hazard between them).
    union { v16bf v; v4i i[2]; } afrag;
    {
      int row = wm * 16 + laneRow;
      int ka  = hiHalf * 8;
      afrag.i[0] = *(const v4i*)&As[buf][row * 40 + ka];
      afrag.i[1] = *(const v4i*)&As[buf][row * 40 + ka + 16];
    }
    union { v16bf v; v4i i[2]; } bfrag[4];
    #pragma unroll
    for (int nt = 0; nt < 4; ++nt) {
      int col = wn * 64 + nt * 16 + laneRow;
      int kb  = hiHalf * 8;
      bfrag[nt].i[0] = *(const v4i*)&Wst[buf][col * 40 + kb];
      bfrag[nt].i[1] = *(const v4i*)&Wst[buf][col * 40 + kb + 16];
    }
    #pragma unroll
    for (int nt = 0; nt < 4; ++nt) {
      acc[nt] = __builtin_amdgcn_wmma_f32_16x16x32_bf16(
          false, afrag.v, false, bfrag[nt].v, (short)0, acc[nt], false, false);
    }
    __syncthreads();
  }

  if constexpr (EPI <= 2) {
    // ---- plain epilogue ----
    const int rowBase = mBlock + wm * 16 + hiHalf * 8;
    #pragma unroll
    for (int nt = 0; nt < 4; ++nt) {
      int col = nBlock + wn * 64 + nt * 16 + laneRow;
      if (col >= N) continue;
      float bv = bias[col];
      #pragma unroll
      for (int i = 0; i < 8; ++i) {
        float v = acc[nt][i] + bv;
        long off = (long)(rowBase + i) * N + col;
        if (EPI == 1) ((unsigned short*)Cptr)[off] = f2bf(v > 0.0f ? v : 0.0f);
        else          ((float*)Cptr)[off] = tanhf(v);
      }
    }
  } else {
    // ---- fused LSTM-cell epilogue ----
    __shared__ __align__(16) float epi[64 * 132];
    __shared__ float zsh[64];
    // spill acc (+bias, gate-interleaved layout) to LDS for cross-wave gather
    #pragma unroll
    for (int nt = 0; nt < 4; ++nt) {
      int colL = wn * 64 + nt * 16 + laneRow;
      float bv = bias[nBlock + colL];
      #pragma unroll
      for (int i = 0; i < 8; ++i)
        epi[(wm * 16 + hiHalf * 8 + i) * 132 + colL] = acc[nt][i] + bv;
    }
    if constexpr (EPI == 3) {
      zsum += __shfl_xor(zsum, 1, 32);
      zsum += __shfl_xor(zsum, 2, 32);
      if ((tid & 3) == 0) zsh[rA] = zsum;
    }
    __syncthreads();

    const int row   = (tid * 8) >> 5;         // 0..63
    const int hbase = (tid * 8) & 31;         // 0,8,16,24
    const long gRow = mBlock + row;
    float z;
    if constexpr (EPI == 3) {
      float zpre = zsh[row] + fa.bz[0];
      float zh = (0.1f * zpre + 1.0f) * 0.5f;
      zh = fminf(0.95f, fmaxf(0.05f, zh));
      z = (fa.noise[gRow] < zh) ? 1.0f : 0.0f;
      if (blockIdx.y == 0 && hbase == 0) fa.z1[gRow] = z;
    } else {
      z = fa.z1[gRow];
    }
    #pragma unroll
    for (int q = 0; q < 8; ++q) {
      int hh = hbase + q;
      float f = sigmoidf_(epi[row * 132 + hh]);
      float i = sigmoidf_(epi[row * 132 + 32 + hh]);
      float o = sigmoidf_(epi[row * 132 + 64 + hh]);
      float g = tanhf(epi[row * 132 + 96 + hh]);
      long idx = gRow * 256 + (blockIdx.y * 32 + hh);
      float cold = fa.c[idx];
      float cn, hn;
      if constexpr (EPI == 3) {
        cn = f * cold + i * g;
      } else {
        cn = (1.0f - z) * cold + z * (f * cold + i * g);
      }
      hn = o * tanhf(cn);
      fa.c[idx] = cn;
      unsigned short hb = f2bf(hn);
      fa.hbf_out[idx] = hb;
      fa.zh_out[idx] = f2bf(z * hn);
      if constexpr (EPI == 4) fa.emb[idx] = hb;
    }
  }
}

// ---------------------------------------------------------------------------
// weight prep: dst[n'][kOff+k] = bf16(src[k*srcStride + n])
// reorder: n' = (h/32)*128 + gate*32 + h%32 with n = gate*256 + h  (gate tiles)
// ---------------------------------------------------------------------------
__global__ __launch_bounds__(256) void prep_wT(
    const float* __restrict__ src, int srcStride,
    unsigned short* __restrict__ dst, int dstStride, int kOff, int K, int N,
    int reorder)
{
  int idx = blockIdx.x * blockDim.x + threadIdx.x;
  if (idx >= N * K) return;
  int n = idx / K, k = idx - n * K;
  int np = n;
  if (reorder) {
    int g = n >> 8, r = n & 255;
    np = (r >> 5) * 128 + g * 32 + (r & 31);
  }
  dst[(long)np * dstStride + kOff + k] = f2bf(src[(long)k * srcStride + n]);
}

__global__ __launch_bounds__(256) void prep_zw(
    const float* __restrict__ src, int srcStride, int col,
    float* __restrict__ dst, int kOff, int K)
{
  int k = blockIdx.x * blockDim.x + threadIdx.x;
  if (k < K) dst[kOff + k] = src[(long)k * srcStride + col];
}

__global__ __launch_bounds__(256) void prep_biasR(
    const float* __restrict__ b, float* __restrict__ out)
{
  int n = blockIdx.x * blockDim.x + threadIdx.x;   // 0..1023
  int g = n >> 8, r = n & 255;
  out[(r >> 5) * 128 + g * 32 + (r & 31)] = b[n];
}

// ---------------------------------------------------------------------------
__global__ __launch_bounds__(256) void init_state(
    float* __restrict__ c1, float* __restrict__ c2,
    unsigned short* __restrict__ h1, unsigned short* __restrict__ h2,
    unsigned short* __restrict__ zh1, unsigned short* __restrict__ zh2,
    float* __restrict__ z1)
{
  int idx = blockIdx.x * blockDim.x + threadIdx.x;   // B*H
  c1[idx] = 0.0f; c2[idx] = 0.0f;
  h1[idx] = 0; h2[idx] = 0; zh1[idx] = 0; zh2[idx] = 0;
  if ((idx & 255) == 0) z1[idx >> 8] = 1.0f;
}

// ---------------------------------------------------------------------------
extern "C" void kernel_launch(void* const* d_in, const int* in_sizes, int n_in,
                              void* d_out, int out_size, void* d_ws, size_t ws_size,
                              hipStream_t stream) {
  const float* x      = (const float*)d_in[0];
  const float* noise1 = (const float*)d_in[1];
  // d_in[2] = noise2 : dead in forward pass
  const float* We1 = (const float*)d_in[3];
  const float* be1 = (const float*)d_in[4];
  const float* We2 = (const float*)d_in[5];
  const float* be2 = (const float*)d_in[6];
  const float* U11_1 = (const float*)d_in[7];
  const float* U21_1 = (const float*)d_in[8];
  const float* W01_1 = (const float*)d_in[9];
  const float* b1    = (const float*)d_in[10];
  const float* U11_2 = (const float*)d_in[11];
  const float* W01_2 = (const float*)d_in[12];
  const float* b2    = (const float*)d_in[13];
  const float* Wh = (const float*)d_in[14];
  const float* bh = (const float*)d_in[15];
  const float* Wp = (const float*)d_in[16];
  const float* bp = (const float*)d_in[17];

  char* ws = (char*)d_ws;
  size_t off = 0;
  auto alloc = [&](size_t bytes) -> void* {
    void* p = ws + off;
    off = (off + bytes + 255) & ~(size_t)255;
    return p;
  };
  unsigned short* WB1  = (unsigned short*)alloc((size_t)1024 * 768 * 2);
  unsigned short* WB2  = (unsigned short*)alloc((size_t)1024 * 512 * 2);
  unsigned short* We1T = (unsigned short*)alloc((size_t)256 * 256 * 2);
  unsigned short* We2T = (unsigned short*)alloc((size_t)256 * 256 * 2);
  unsigned short* WhT  = (unsigned short*)alloc((size_t)256 * 256 * 2);
  unsigned short* WpT  = (unsigned short*)alloc((size_t)64 * 256 * 2);
  float* zw1    = (float*)alloc(768 * 4);
  float* biasR1 = (float*)alloc(1024 * 4);
  float* biasR2 = (float*)alloc(1024 * 4);
  float* c1     = (float*)alloc((size_t)Bsz * HH * 4);
  float* c2     = (float*)alloc((size_t)Bsz * HH * 4);
  unsigned short* h1A = (unsigned short*)alloc((size_t)Bsz * HH * 2);
  unsigned short* h1B = (unsigned short*)alloc((size_t)Bsz * HH * 2);
  unsigned short* h2A = (unsigned short*)alloc((size_t)Bsz * HH * 2);
  unsigned short* h2B = (unsigned short*)alloc((size_t)Bsz * HH * 2);
  unsigned short* zh1 = (unsigned short*)alloc((size_t)Bsz * HH * 2);
  unsigned short* zh2 = (unsigned short*)alloc((size_t)Bsz * HH * 2);
  float* z1    = (float*)alloc(Bsz * 4);
  unsigned short* xe  = (unsigned short*)alloc((size_t)TB * HH * 2);
  unsigned short* tmp = (unsigned short*)alloc((size_t)TB * HH * 2);  // enc mid / hid
  unsigned short* emb = (unsigned short*)alloc((size_t)TB * HH * 2);

  // ---- weight prep ----
  prep_wT<<<256, 256, 0, stream>>>(We1, 256, We1T, 256, 0, 256, 256, 0);
  prep_wT<<<256, 256, 0, stream>>>(We2, 256, We2T, 256, 0, 256, 256, 0);
  prep_wT<<<256, 256, 0, stream>>>(Wh,  256, WhT,  256, 0, 256, 256, 0);
  prep_wT<<<64,  256, 0, stream>>>(Wp,  64,  WpT,  256, 0, 256, 64, 0);
  prep_wT<<<1024, 256, 0, stream>>>(W01_1, 1025, WB1, 768, 0,   256, 1024, 1);
  prep_wT<<<1024, 256, 0, stream>>>(U21_1, 1025, WB1, 768, 256, 256, 1024, 1);
  prep_wT<<<1024, 256, 0, stream>>>(U11_1, 1025, WB1, 768, 512, 256, 1024, 1);
  prep_wT<<<1024, 256, 0, stream>>>(W01_2, 1025, WB2, 512, 0,   256, 1024, 1);
  prep_wT<<<1024, 256, 0, stream>>>(U11_2, 1025, WB2, 512, 256, 256, 1024, 1);
  prep_zw<<<1, 256, 0, stream>>>(W01_1, 1025, 1024, zw1, 0,   256);
  prep_zw<<<1, 256, 0, stream>>>(U21_1, 1025, 1024, zw1, 256, 256);
  prep_zw<<<1, 256, 0, stream>>>(U11_1, 1025, 1024, zw1, 512, 256);
  prep_biasR<<<4, 256, 0, stream>>>(b1, biasR1);
  prep_biasR<<<4, 256, 0, stream>>>(b2, biasR2);

  FuseArgs none = {};

  // ---- encoder ----
  dim3 ge(TB / 64, 2);
  gemm_wmma<1, true ><<<ge, 256, 0, stream>>>(x,   nullptr, nullptr, We1T, be1, tmp, TB, 256, 256, none);
  gemm_wmma<1, false><<<ge, 256, 0, stream>>>(tmp, nullptr, nullptr, We2T, be2, xe,  TB, 256, 256, none);

  // ---- recurrent init ----
  init_state<<<(Bsz * HH) / 256, 256, 0, stream>>>(c1, c2, h1A, h2A, zh1, zh2, z1);

  // ---- sequential scan: 2 fused kernels per step ----
  dim3 gr(1024 / 64, 1024 / 128);
  for (int t = 0; t < Tt; ++t) {
    unsigned short* h1_in  = (t & 1) ? h1B : h1A;
    unsigned short* h1_out = (t & 1) ? h1A : h1B;
    unsigned short* h2_in  = (t & 1) ? h2B : h2A;
    unsigned short* h2_out = (t & 1) ? h2A : h2B;

    FuseArgs fa1 = {};
    fa1.c = c1; fa1.hbf_out = h1_out; fa1.zh_out = zh1; fa1.z1 = z1;
    fa1.noise = noise1 + (long)t * Bsz; fa1.zw = zw1; fa1.bz = b1 + 1024;
    gemm_wmma<3, false><<<gr, 256, 0, stream>>>(
        h1_in, zh2, xe + (long)t * Bsz * HH, WB1, biasR1, nullptr,
        1024, 1024, 768, fa1);

    FuseArgs fa2 = {};
    fa2.c = c2; fa2.hbf_out = h2_out; fa2.zh_out = zh2; fa2.z1 = z1;
    fa2.emb = emb + (long)t * Bsz * HH;
    gemm_wmma<4, false><<<gr, 256, 0, stream>>>(
        h2_in, zh1, nullptr, WB2, biasR2, nullptr,
        1024, 1024, 512, fa2);
  }

  // ---- head ----
  gemm_wmma<1, false><<<ge, 256, 0, stream>>>(emb, nullptr, nullptr, WhT, bh, tmp, TB, 256, 256, none);
  dim3 gp(TB / 64, 1);
  gemm_wmma<2, false><<<gp, 256, 0, stream>>>(tmp, nullptr, nullptr, WpT, bp, (float*)d_out, TB, 64, 256, none);
}
